// NeuralModel_55791625175293
// MI455X (gfx1250) — matrix operand order repaired and checked by reading, not compiled, not run on hardware
//
#include <hip/hip_runtime.h>

typedef __attribute__((ext_vector_type(16))) _Float16 v16h;
typedef __attribute__((ext_vector_type(8)))  float    v8f;

#define NPTS 1048576
#define NF   4
#define QG   64
#define CCH  4
#define BLK  256
#define OUTD 26

__global__ __launch_bounds__(BLK) void qff_nerf_kernel(
    const float* __restrict__ points, const float* __restrict__ dirs,
    const float* __restrict__ grid,
    const float* __restrict__ Wg1, const float* __restrict__ Wg2,
    const float* __restrict__ density_bias,
    const float* __restrict__ Wc1, const float* __restrict__ Wc2,
    const float* __restrict__ Wn1, const float* __restrict__ Wn2,
    const float* __restrict__ Ws1, const float* __restrict__ Ws2,
    float* __restrict__ out)
{
    // ---- LDS staging (≈57 KB / workgroup, well under 320 KB WGP budget) ----
    __shared__ _Float16 sFeats[BLK * 32];   // per-point QFF features (f16, WMMA A source)
    __shared__ _Float16 sHid  [BLK * 32];   // relu(layer1), K padded 16->32 with zeros
    __shared__ float    sS    [BLK * 16];   // layer2 output rows
    __shared__ _Float16 sWgB1[16 * 32];     // Wg1 col-major [n][k]  (WMMA B frag source)
    __shared__ _Float16 sWgB2[16 * 32];     // Wg2 col-major, k>=16 zero
    __shared__ float    sWg1f[32 * 16];     // Wg1 row-major float (for VJP matvec)
    __shared__ float    sWg2c0[16];         // Wg2[:,0]
    __shared__ float    sWc1[18 * 16];
    __shared__ float    sWc2[16 * 3];
    __shared__ float    sWn1[15 * 16];
    __shared__ float    sWn2[16 * 3];
    __shared__ float    sWs1[15 * 16];
    __shared__ float    sWs2[16 * 16];

    const int tid = threadIdx.x;

    for (int i = tid; i < 512; i += BLK) {
        const int k = i & 31, n = i >> 5;
        sWgB1[n * 32 + k] = (_Float16)Wg1[k * 16 + n];
        sWgB2[n * 32 + k] = (k < 16) ? (_Float16)Wg2[k * 16 + n] : (_Float16)0.0f;
        sWg1f[i] = Wg1[i];
    }
    for (int i = tid; i < 16;      i += BLK) sWg2c0[i] = Wg2[i * 16];
    for (int i = tid; i < 18 * 16; i += BLK) sWc1[i] = Wc1[i];
    for (int i = tid; i < 16 * 3;  i += BLK) sWc2[i] = Wc2[i];
    for (int i = tid; i < 15 * 16; i += BLK) sWn1[i] = Wn1[i];
    for (int i = tid; i < 16 * 3;  i += BLK) sWn2[i] = Wn2[i];
    for (int i = tid; i < 15 * 16; i += BLK) sWs1[i] = Ws1[i];
    for (int i = tid; i < 16 * 16; i += BLK) sWs2[i] = Ws2[i];
    for (int k = 16; k < 32; ++k) sHid[tid * 32 + k] = (_Float16)0.0f; // K pad
    __syncthreads();

    const int   gpt = blockIdx.x * BLK + tid;          // one thread = one point
    const float px = points[gpt * 3 + 0];
    const float py = points[gpt * 3 + 1];
    const float pz = points[gpt * 3 + 2];

    const float freqs[NF] = {1.0f, 3.17480210393640f, 10.0793683991589f, 32.0f};

    // ---- Phase 1: QFF forward encode (64 L2-resident gathers) -> sFeats ----
    for (int f = 0; f < NF; ++f) {
        const float freq = freqs[f];
        float sn[3], cs[3];
        sn[0] = __sinf(px * freq); cs[0] = __cosf(px * freq);
        sn[1] = __sinf(py * freq); cs[1] = __cosf(py * freq);
        sn[2] = __sinf(pz * freq); cs[2] = __cosf(pz * freq);
        for (int s = 0; s < 2; ++s) {
            float w[3]; int i0[3];
            for (int k = 0; k < 3; ++k) {
                const float sc = s ? cs[k] : sn[k];
                const float u  = (sc + 1.0f) * 31.5f;
                int ii = (int)floorf(u);
                ii = ii < 0 ? 0 : (ii > QG - 2 ? QG - 2 : ii);
                i0[k] = ii; w[k] = u - (float)ii;
            }
            const float* gbase = grid +
                ((((size_t)(f * 2 + s) * QG + i0[0]) * QG + i0[1]) * QG + i0[2]) * CCH;
            float a0 = 0.f, a1 = 0.f, a2 = 0.f, a3 = 0.f;
            #pragma unroll
            for (int corner = 0; corner < 8; ++corner) {
                const int a = (corner >> 2) & 1, b = (corner >> 1) & 1, c = corner & 1;
                const float wt = (a ? w[0] : 1.0f - w[0]) *
                                 (b ? w[1] : 1.0f - w[1]) *
                                 (c ? w[2] : 1.0f - w[2]);
                const float4 v = *(const float4*)(gbase +
                    (size_t)a * QG * QG * CCH + b * QG * CCH + c * CCH);
                a0 += wt * v.x; a1 += wt * v.y; a2 += wt * v.z; a3 += wt * v.w;
            }
            const int fo = (f * 2 + s) * 4;
            sFeats[tid * 32 + fo + 0] = (_Float16)a0;
            sFeats[tid * 32 + fo + 1] = (_Float16)a1;
            sFeats[tid * 32 + fo + 2] = (_Float16)a2;
            sFeats[tid * 32 + fo + 3] = (_Float16)a3;
        }
    }

    // ---- Phase 2: geometry MLP via v_wmma_f32_16x16x32_f16 (intra-wave) ----
    const int lane     = tid & 31;
    const int waveBase = tid & ~31;           // this wave's 32 local points
    const int nIdx     = lane & 15;           // D column / B column
    const int kbA      = (lane < 16) ? 0 : 8; // A frag: K offsets per ISA layout
    const int kbB      = (lane < 16) ? 0 : 8; // B frag: same K offsets as A frag
    const int mb       = (lane < 16) ? 0 : 8; // D rows r+8*(lane>=16)

    v16h bfrag1, bfrag2;
    #pragma unroll
    for (int t = 0; t < 8; ++t) {
        bfrag1[t]     = sWgB1[nIdx * 32 + kbB + t];
        bfrag1[t + 8] = sWgB1[nIdx * 32 + 16 + kbB + t];
        bfrag2[t]     = sWgB2[nIdx * 32 + kbB + t];
        bfrag2[t + 8] = sWgB2[nIdx * 32 + 16 + kbB + t];
    }

    #pragma unroll
    for (int tile = 0; tile < 2; ++tile) {
        const int tbase = waveBase + tile * 16;
        const int prow  = tbase + (lane & 15);
        v16h afrag;
        #pragma unroll
        for (int t = 0; t < 8; ++t) {
            afrag[t]     = sFeats[prow * 32 + kbA + t];
            afrag[t + 8] = sFeats[prow * 32 + 16 + kbA + t];
        }
        v8f acc = {0.f, 0.f, 0.f, 0.f, 0.f, 0.f, 0.f, 0.f};
        acc = __builtin_amdgcn_wmma_f32_16x16x32_f16(
                  false, afrag, false, bfrag1, (short)0, acc, false, false);
        #pragma unroll
        for (int r = 0; r < 8; ++r)
            sHid[(tbase + mb + r) * 32 + nIdx] = (_Float16)fmaxf(acc[r], 0.0f);

        v16h afrag2;
        #pragma unroll
        for (int t = 0; t < 8; ++t) {
            afrag2[t]     = sHid[prow * 32 + kbA + t];
            afrag2[t + 8] = sHid[prow * 32 + 16 + kbA + t];   // zero pad region
        }
        v8f acc2 = {0.f, 0.f, 0.f, 0.f, 0.f, 0.f, 0.f, 0.f};
        acc2 = __builtin_amdgcn_wmma_f32_16x16x32_f16(
                   false, afrag2, false, bfrag2, (short)0, acc2, false, false);
        #pragma unroll
        for (int r = 0; r < 8; ++r)
            sS[(tbase + mb + r) * 16 + nIdx] = acc2[r];
    }
    __syncthreads();

    // ---- Phase 3: density + VJP seed dfeat = Wg1 @ (relu' ⊙ Wg2[:,0]) ----
    float sarr[16];
    #pragma unroll
    for (int j = 0; j < 16; ++j) sarr[j] = sS[tid * 16 + j];
    const float density = expf(sarr[0] + density_bias[0]);

    float m[16];
    #pragma unroll
    for (int j = 0; j < 16; ++j)
        m[j] = ((float)sHid[tid * 32 + j] > 0.0f) ? sWg2c0[j] : 0.0f;
    float dfeat[32];
    #pragma unroll
    for (int i = 0; i < 32; ++i) {
        float a = 0.f;
        #pragma unroll
        for (int j = 0; j < 16; ++j) a += sWg1f[i * 16 + j] * m[j];
        dfeat[i] = a;
    }

    // ---- Phase 4: backward through encoder (recomputed gathers) -> gn ----
    float g0 = 0.f, g1 = 0.f, g2 = 0.f;
    for (int f = 0; f < NF; ++f) {
        const float freq = freqs[f];
        float sn[3], cs[3];
        sn[0] = __sinf(px * freq); cs[0] = __cosf(px * freq);
        sn[1] = __sinf(py * freq); cs[1] = __cosf(py * freq);
        sn[2] = __sinf(pz * freq); cs[2] = __cosf(pz * freq);
        for (int s = 0; s < 2; ++s) {
            float w[3], dud[3]; int i0[3];
            for (int k = 0; k < 3; ++k) {
                const float sc = s ? cs[k] : sn[k];
                const float u  = (sc + 1.0f) * 31.5f;
                int ii = (int)floorf(u);
                ii = ii < 0 ? 0 : (ii > QG - 2 ? QG - 2 : ii);
                i0[k] = ii; w[k] = u - (float)ii;
                dud[k] = 31.5f * freq * (s ? -sn[k] : cs[k]);  // d(u_k)/d(p_k)
            }
            const float* gbase = grid +
                ((((size_t)(f * 2 + s) * QG + i0[0]) * QG + i0[1]) * QG + i0[2]) * CCH;
            const float* dfp = &dfeat[(f * 2 + s) * 4];
            float l0 = 0.f, l1 = 0.f, l2 = 0.f;
            #pragma unroll
            for (int corner = 0; corner < 8; ++corner) {
                const int a = (corner >> 2) & 1, b = (corner >> 1) & 1, c = corner & 1;
                const float4 v = *(const float4*)(gbase +
                    (size_t)a * QG * QG * CCH + b * QG * CCH + c * CCH);
                const float dot = v.x * dfp[0] + v.y * dfp[1] + v.z * dfp[2] + v.w * dfp[3];
                const float wa = a ? w[0] : 1.f - w[0];
                const float wb = b ? w[1] : 1.f - w[1];
                const float wc = c ? w[2] : 1.f - w[2];
                l0 += (a ? 1.f : -1.f) * wb * wc * dot;
                l1 += (b ? 1.f : -1.f) * wa * wc * dot;
                l2 += (c ? 1.f : -1.f) * wa * wb * dot;
            }
            g0 += l0 * dud[0]; g1 += l1 * dud[1]; g2 += l2 * dud[2];
        }
    }
    const float gpx = density * g0, gpy = density * g1, gpz = density * g2;
    const float gnorm = fmaxf(sqrtf(gpx * gpx + gpy * gpy + gpz * gpz), 1e-8f);
    const float gnx = -gpx / gnorm, gny = -gpy / gnorm, gnz = -gpz / gnorm;

    // ---- Phase 5: rgb / normal / seg heads (tiny VALU matvecs) ----
    float h[15];
    #pragma unroll
    for (int j = 0; j < 15; ++j) h[j] = sarr[j + 1];
    const float dx = dirs[gpt * 3 + 0], dy = dirs[gpt * 3 + 1], dz = dirs[gpt * 3 + 2];

    float in18[18];
    #pragma unroll
    for (int j = 0; j < 15; ++j) in18[j] = h[j];
    in18[15] = dx; in18[16] = dy; in18[17] = dz;
    float hc[16];
    #pragma unroll
    for (int j = 0; j < 16; ++j) {
        float a = 0.f;
        for (int i = 0; i < 18; ++i) a += in18[i] * sWc1[i * 16 + j];
        hc[j] = fmaxf(a, 0.f);
    }
    float rgb[3];
    #pragma unroll
    for (int j = 0; j < 3; ++j) {
        float a = 0.f;
        for (int i = 0; i < 16; ++i) a += hc[i] * sWc2[i * 3 + j];
        rgb[j] = 1.f / (1.f + expf(-a));
    }

    float hn[16];
    #pragma unroll
    for (int j = 0; j < 16; ++j) {
        float a = 0.f;
        for (int i = 0; i < 15; ++i) a += h[i] * sWn1[i * 16 + j];
        hn[j] = fmaxf(a, 0.f);
    }
    float nv[3];
    #pragma unroll
    for (int j = 0; j < 3; ++j) {
        float a = 0.f;
        for (int i = 0; i < 16; ++i) a += hn[i] * sWn2[i * 3 + j];
        nv[j] = a;
    }
    const float nn = fmaxf(sqrtf(nv[0] * nv[0] + nv[1] * nv[1] + nv[2] * nv[2]), 1e-12f);
    nv[0] /= nn; nv[1] /= nn; nv[2] /= nn;

    float hs[16];
    #pragma unroll
    for (int j = 0; j < 16; ++j) {
        float a = 0.f;
        for (int i = 0; i < 15; ++i) a += h[i] * sWs1[i * 16 + j];
        hs[j] = fmaxf(a, 0.f);
    }
    float seg[16];
    #pragma unroll
    for (int j = 0; j < 16; ++j) {
        float a = 0.f;
        for (int i = 0; i < 16; ++i) a += hs[i] * sWs2[i * 16 + j];
        seg[j] = a;
    }

    // ---- Phase 6: write [rgb(3), density(1), gn(3), n(3), seg(16)] ----
    float* o = out + (size_t)gpt * OUTD;
    o[0] = rgb[0]; o[1] = rgb[1]; o[2] = rgb[2];
    o[3] = density;
    o[4] = gnx; o[5] = gny; o[6] = gnz;
    o[7] = nv[0]; o[8] = nv[1]; o[9] = nv[2];
    #pragma unroll
    for (int j = 0; j < 16; ++j) o[10 + j] = seg[j];
}

extern "C" void kernel_launch(void* const* d_in, const int* in_sizes, int n_in,
                              void* d_out, int out_size, void* d_ws, size_t ws_size,
                              hipStream_t stream) {
    (void)in_sizes; (void)n_in; (void)d_ws; (void)ws_size; (void)out_size;
    qff_nerf_kernel<<<NPTS / BLK, BLK, 0, stream>>>(
        (const float*)d_in[0],  // points
        (const float*)d_in[1],  // dirs
        (const float*)d_in[2],  // grid
        (const float*)d_in[3],  // Wg1
        (const float*)d_in[4],  // Wg2
        (const float*)d_in[5],  // density_bias
        (const float*)d_in[6],  // Wc1
        (const float*)d_in[7],  // Wc2
        (const float*)d_in[8],  // Wn1
        (const float*)d_in[9],  // Wn2
        (const float*)d_in[10], // Ws1
        (const float*)d_in[11], // Ws2
        (float*)d_out);
}
